// EntangledInterferenceLayer_5523327943178
// MI455X (gfx1250) — matrix operand order repaired and checked by reading, not compiled, not run on hardware
//
#include <hip/hip_runtime.h>
#include <hip/hip_bf16.h>
#include <math.h>

// ---------------------------------------------------------------------------
// Types for CDNA5 WMMA (bf16 in, f32 accumulate)
// ---------------------------------------------------------------------------
typedef __bf16 bf16_t;
typedef __attribute__((ext_vector_type(16))) __bf16 v16bf;
typedef __attribute__((ext_vector_type(8)))  __bf16 v8bf;
typedef __attribute__((ext_vector_type(4)))  __bf16 v4bf;
typedef __attribute__((ext_vector_type(8)))  float   v8f;
typedef __attribute__((ext_vector_type(4)))  unsigned int u32x4;
typedef __attribute__((ext_vector_type(8)))  int i32x8;
typedef __attribute__((ext_vector_type(4)))  int i32x4;
typedef __attribute__((ext_vector_type(4)))  int v4i;

#define BB   4
#define LL   1024
#define DD   1024
#define HH   16
#define HDIM 64
#define MTOK 4096              // B*L
#define LDT  40                // LDS row stride in bf16 elems (80B, 16B aligned)
#define SCALE_QK 0.125f        // 1/sqrt(64)
#define NOISE_P 0.05f
#define NOISE_S 0.1f

// ---------------------------------------------------------------------------
// CDNA5 feature gates (fall back to manual LDS staging when missing)
// ---------------------------------------------------------------------------
#if defined(__has_builtin)
#if __has_builtin(__builtin_amdgcn_global_load_async_to_lds_b128)
#define USE_ASYNC_LDS 1
#endif
#if __has_builtin(__builtin_amdgcn_tensor_load_to_lds) && \
    __has_builtin(__builtin_amdgcn_s_wait_tensorcnt)
#define USE_TDM 1
#endif
#endif

// async-to-LDS builtin wants v4i pointers (per compiler diagnostic)
typedef __attribute__((address_space(1))) v4i as1_v4i;
typedef __attribute__((address_space(3))) v4i as3_v4i;

__device__ __forceinline__ as1_v4i* gptr_v4i(const void* p) {
  return (as1_v4i*)(unsigned long long)(size_t)p;
}
__device__ __forceinline__ as3_v4i* lptr_v4i(const void* p) {
  // generic LDS address: low 32 bits are the LDS byte offset (ISA aperture rule)
  return (as3_v4i*)(unsigned)(size_t)p;
}
__device__ __forceinline__ unsigned lds_offset(const void* p) {
  return (unsigned)(size_t)p;
}

__device__ __forceinline__ void wait_async_lds() {
#if defined(USE_ASYNC_LDS)
#if __has_builtin(__builtin_amdgcn_s_wait_asynccnt)
  __builtin_amdgcn_s_wait_asynccnt(0);
#else
  asm volatile("s_wait_asynccnt 0x0" ::: "memory");
#endif
#endif
}

#if defined(USE_TDM)
// 2D TDM tile load: tile_d1 rows x tile_d0 elems (2-byte data), LDS rows padded
// 64B data + 16B pad (pad_interval=16 DWORDs -> enc 3, pad_amount=4 DWORDs -> enc 3)
__device__ __forceinline__ void tdm_load_tile_2d(
    unsigned lds_addr, const void* gaddr,
    unsigned tensor_d0, unsigned tensor_d1,
    unsigned tile_d0, unsigned tile_d1,
    unsigned long long stride0_elems) {
  unsigned long long ga = (unsigned long long)(size_t)gaddr;
  u32x4 g0;
  g0[0] = 1u;                                            // count=1 (valid user D#)
  g0[1] = lds_addr;                                      // lds_addr (bytes)
  g0[2] = (unsigned)(ga & 0xFFFFFFFFu);                  // global_addr[31:0]
  g0[3] = (unsigned)((ga >> 32) & 0x1FFFFFFu) | (2u << 30);  // [56:32] | type=2

  unsigned long long q0 = 0ull, q1 = 0ull, q2 = 0ull, q3 = 0ull;
  q0 |= (1ull << 16);                                    // data_size = 2 bytes
  q0 |= (1ull << 20);                                    // pad_enable
  q0 |= (3ull << 22);                                    // pad_interval = 16 DW
  q0 |= (3ull << 25);                                    // pad_amount   = 4 DW
  q0 |= ((unsigned long long)(tensor_d0 & 0xFFFFu)) << 48;  // tensor_dim0[15:0]
  q1 |= (unsigned long long)(tensor_d0 >> 16);              // tensor_dim0[31:16]
  q1 |= ((unsigned long long)tensor_d1) << 16;              // tensor_dim1
  q1 |= ((unsigned long long)(tile_d0 & 0xFFFFu)) << 48;    // tile_dim0
  q2 |= (unsigned long long)(tile_d1 & 0xFFFFu);            // tile_dim1
  q2 |= (stride0_elems & 0xFFFFFFFFull) << 32;              // dim0_stride[31:0]
  q3 |= (stride0_elems >> 32) & 0xFFFFull;                  // dim0_stride[47:32]

  i32x8 g1;
  g1[0] = (int)(unsigned)q0;  g1[1] = (int)(unsigned)(q0 >> 32);
  g1[2] = (int)(unsigned)q1;  g1[3] = (int)(unsigned)(q1 >> 32);
  g1[4] = (int)(unsigned)q2;  g1[5] = (int)(unsigned)(q2 >> 32);
  g1[6] = (int)(unsigned)q3;  g1[7] = (int)(unsigned)(q3 >> 32);
  i32x4 z4 = {};
#if __clang_major__ >= 23
  i32x8 z8 = {};
  __builtin_amdgcn_tensor_load_to_lds(g0, g1, z4, z4, z8, 0);
#else
  __builtin_amdgcn_tensor_load_to_lds(g0, g1, z4, z4, 0);
#endif
}
#endif  // USE_TDM

// ---------------------------------------------------------------------------
// Deterministic counter-based noise (same masks for real/imag partners)
// ---------------------------------------------------------------------------
__device__ __forceinline__ unsigned pcg_hash(unsigned v) {
  v = v * 747796405u + 2891336453u;
  v = ((v >> ((v >> 28u) + 4u)) ^ v) * 277803737u;
  return (v >> 22u) ^ v;
}
__device__ __forceinline__ float u01(unsigned h) {
  return (float)(h >> 8) * (1.0f / 16777216.0f);
}
__device__ __forceinline__ float apply_noise(float v, unsigned nid, unsigned imag, unsigned idx) {
  unsigned base = pcg_hash(idx ^ (nid * 0x9E3779B9u));
  float flip = (u01(pcg_hash(base ^ 0xA511E9B3u)) < NOISE_P) ? -1.0f : 1.0f;
  float ampm = (u01(pcg_hash(base ^ 0x63D83595u)) < NOISE_P) ? 1.0f : 0.0f;
  unsigned g1 = pcg_hash(base ^ (0x7F4A7C15u + imag));
  unsigned g2 = pcg_hash(base ^ (0x94D049BBu + imag));
  float uu = fmaxf(u01(g1), 1.0e-7f);
  float g  = sqrtf(-2.0f * __logf(uu)) * __cosf(6.2831853f * u01(g2));
  float keep = (u01(pcg_hash(base ^ 0xB5297A4Du)) < NOISE_P * 0.5f) ? 0.0f : 1.0f;
  return (v * flip + g * NOISE_S * ampm) * keep;
}

// ---------------------------------------------------------------------------
// WMMA fragment gathers from LDS (per documented CDNA5 VGPR layouts)
// ---------------------------------------------------------------------------
__device__ __forceinline__ v16bf frag_rowK(const bf16_t* p) {
  v8bf lo = *(const v8bf*)(p);
  v8bf hi = *(const v8bf*)(p + 16);
  return __builtin_shufflevector(lo, hi, 0,1,2,3,4,5,6,7,8,9,10,11,12,13,14,15);
}
__device__ __forceinline__ v16bf frag_colK(const bf16_t* p) {
  v8bf lo = *(const v8bf*)(p);
  v8bf hi = *(const v8bf*)(p + 16);
  return __builtin_shufflevector(lo, hi, 0,1,2,3,4,5,6,7,8,9,10,11,12,13,14,15);
}

// Stage one 128x32 bf16 tile: global (row-major, ld=ldsrc elems) -> padded LDS
__device__ __forceinline__ void stage_tile_async_or_copy(
    bf16_t* __restrict__ lds, const bf16_t* __restrict__ g, int ldsrc,
    int frow, int fseg) {
  const bf16_t* gp = g + (size_t)frow * ldsrc + fseg;
  bf16_t* lp = lds + frow * LDT + fseg;
#if defined(USE_ASYNC_LDS)
  __builtin_amdgcn_global_load_async_to_lds_b128(gptr_v4i(gp),     lptr_v4i(lp),     0, 0);
  __builtin_amdgcn_global_load_async_to_lds_b128(gptr_v4i(gp + 8), lptr_v4i(lp + 8), 0, 0);
#else
  const uint4* gv = (const uint4*)gp;
  uint4 a0 = gv[0], a1 = gv[1];
  *(uint4*)(lp)     = a0;
  *(uint4*)(lp + 8) = a1;
#endif
}

// ---------------------------------------------------------------------------
// f32 -> bf16 pack
// ---------------------------------------------------------------------------
__global__ __launch_bounds__(256) void pack_bf16_kernel(
    const float* __restrict__ s, bf16_t* __restrict__ d, int n) {
  int i = (blockIdx.x * 256 + threadIdx.x) * 4;
  if (i >= n) return;
  float4 f = *(const float4*)(s + i);
  v4bf o;
  o[0] = (bf16_t)f.x; o[1] = (bf16_t)f.y; o[2] = (bf16_t)f.z; o[3] = (bf16_t)f.w;
  *(v4bf*)(d + i) = o;
}

// ---------------------------------------------------------------------------
// Tiled WMMA GEMM: C[M,N] = A[M,K](bf16) * Bw[N,K](bf16)^T + bias
// 256 thr = 8 waves (4m x 2n); wave tile 32x64 = 2x4 WMMA 16x16 tiles; BK=32
// A tile: per-lane async-to-LDS (ASYNCcnt). B tile: TDM tensor_load_to_lds
// (TENSORcnt, wave0-issued) when available.
// MODE 0: store f32. MODE 1: fused noise (+ ent diag per head) -> bf16.
// ---------------------------------------------------------------------------
template <int MODE>
__global__ __launch_bounds__(256) void gemm_bf16_kernel(
    const bf16_t* __restrict__ A, const bf16_t* __restrict__ Bw,
    const float* __restrict__ bias, int M, int N, int K,
    int nid, int imag, const float* __restrict__ ent,
    bf16_t* __restrict__ out_bf, float* __restrict__ out_f) {
  __shared__ alignas(16) bf16_t As[128 * LDT];
  __shared__ alignas(16) bf16_t Bs[128 * LDT];

  const int m0 = blockIdx.y * 128, n0 = blockIdx.x * 128;
  const int lane = threadIdx.x & 31, wid = threadIdx.x >> 5;
  const int wm = (wid & 3) * 32, wn = (wid >> 2) * 64;
  const int frow = threadIdx.x >> 1, fseg = (threadIdx.x & 1) << 4;

  v8f acc[2][4] = {};

  for (int k0 = 0; k0 < K; k0 += 32) {
    if (k0 + 32 < K) {
      __builtin_prefetch(A  + (size_t)(m0 + frow) * K + k0 + 32, 0, 1);
      __builtin_prefetch(Bw + (size_t)(n0 + frow) * K + k0 + 32, 0, 1);
    }
    // A tile: per-lane async copy
    stage_tile_async_or_copy(As, A + (size_t)m0 * K + k0, K, frow, fseg);
    // B tile: TDM bulk tile load (one issue per workgroup), else per-lane path
#if defined(USE_TDM)
    if (wid == 0) {
      tdm_load_tile_2d(lds_offset(Bs),
                       Bw + (size_t)n0 * K + k0,
                       (unsigned)K, 128u, 32u, 128u, (unsigned long long)K);
      __builtin_amdgcn_s_wait_tensorcnt(0);
    }
#else
    stage_tile_async_or_copy(Bs, Bw + (size_t)n0 * K + k0, K, frow, fseg);
#endif
    wait_async_lds();
    __syncthreads();

    const int arow = lane & 15, koff = (lane < 16) ? 0 : 8;
    const int kb = (lane < 16) ? 0 : 8;
    v16bf af[2], bq[4];
#pragma unroll
    for (int mt = 0; mt < 2; ++mt)
      af[mt] = frag_rowK(&As[(wm + mt * 16 + arow) * LDT + koff]);
#pragma unroll
    for (int nt = 0; nt < 4; ++nt)
      bq[nt] = frag_colK(&Bs[(wn + nt * 16 + (lane & 15)) * LDT + kb]);
#pragma unroll
    for (int mt = 0; mt < 2; ++mt)
#pragma unroll
      for (int nt = 0; nt < 4; ++nt)
        acc[mt][nt] = __builtin_amdgcn_wmma_f32_16x16x32_bf16(
            false, af[mt], false, bq[nt], (short)0, acc[mt][nt], false, false);
    __syncthreads();
  }

#pragma unroll
  for (int mt = 0; mt < 2; ++mt) {
#pragma unroll
    for (int nt = 0; nt < 4; ++nt) {
      const int n  = n0 + wn + nt * 16 + (lane & 15);
      const int mb = m0 + wm + mt * 16 + ((lane < 16) ? 0 : 8);
      const float bia = bias[n];
      float entd = 1.0f;
      if (MODE == 1 && ent != nullptr) { int h = n >> 6; entd = ent[h * HH + h]; }
#pragma unroll
      for (int i = 0; i < 8; ++i) {
        const int m = mb + i;
        float v = acc[mt][nt][i] + bia;
        if (MODE == 1) {
          unsigned idx = (unsigned)m * (unsigned)N + (unsigned)n;
          v = apply_noise(v, (unsigned)nid, (unsigned)imag, idx) * entd;
          out_bf[(size_t)m * N + n] = (bf16_t)v;
        } else {
          out_f[(size_t)m * N + n] = v;
        }
      }
    }
  }
}

// ---------------------------------------------------------------------------
// Scores: S[b,h] = (Qr Kr^T + Qi Ki^T) * SCALE, masked -> attn (f32, d_out)
// ---------------------------------------------------------------------------
__global__ __launch_bounds__(256) void scores_kernel(
    const bf16_t* __restrict__ qr, const bf16_t* __restrict__ qi,
    const bf16_t* __restrict__ kr, const bf16_t* __restrict__ ki,
    const int* __restrict__ mask, float* __restrict__ attn) {
  __shared__ alignas(16) bf16_t As[128 * LDT];
  __shared__ alignas(16) bf16_t Bs[128 * LDT];

  const int b = blockIdx.z >> 4, h = blockIdx.z & 15;
  const size_t base = (size_t)b * LL * DD + (size_t)h * HDIM;
  const bf16_t* Ap[2] = { qr + base, qi + base };
  const bf16_t* Bp[2] = { kr + base, ki + base };

  const int m0 = blockIdx.y * 128, n0 = blockIdx.x * 128;
  const int lane = threadIdx.x & 31, wid = threadIdx.x >> 5;
  const int wm = (wid & 3) * 32, wn = (wid >> 2) * 64;
  const int frow = threadIdx.x >> 1, fseg = (threadIdx.x & 1) << 4;

  v8f acc[2][4] = {};

  for (int pr = 0; pr < 2; ++pr) {
    for (int kh = 0; kh < 2; ++kh) {
      stage_tile_async_or_copy(As, Ap[pr] + (size_t)m0 * DD + kh * 32, DD, frow, fseg);
      stage_tile_async_or_copy(Bs, Bp[pr] + (size_t)n0 * DD + kh * 32, DD, frow, fseg);
      wait_async_lds();
      __syncthreads();
      const int arow = lane & 15, koff = (lane < 16) ? 0 : 8;
      const int kb = (lane < 16) ? 0 : 8;
      v16bf af[2], bq[4];
#pragma unroll
      for (int mt = 0; mt < 2; ++mt)
        af[mt] = frag_rowK(&As[(wm + mt * 16 + arow) * LDT + koff]);
#pragma unroll
      for (int nt = 0; nt < 4; ++nt)
        bq[nt] = frag_colK(&Bs[(wn + nt * 16 + (lane & 15)) * LDT + kb]);
#pragma unroll
      for (int mt = 0; mt < 2; ++mt)
#pragma unroll
        for (int nt = 0; nt < 4; ++nt)
          acc[mt][nt] = __builtin_amdgcn_wmma_f32_16x16x32_bf16(
              false, af[mt], false, bq[nt], (short)0, acc[mt][nt], false, false);
      __syncthreads();
    }
  }

  float* S = attn + (size_t)(b * HH + h) * LL * LL;
#pragma unroll
  for (int mt = 0; mt < 2; ++mt) {
#pragma unroll
    for (int nt = 0; nt < 4; ++nt) {
      const int n  = n0 + wn + nt * 16 + (lane & 15);
      const int mb = m0 + wm + mt * 16 + ((lane < 16) ? 0 : 8);
      const int mk = mask[b * LL + n];
#pragma unroll
      for (int i = 0; i < 8; ++i) {
        float v = acc[mt][nt][i] * SCALE_QK;
        if (mk == 0) v = -1.0e9f;
        S[(size_t)(mb + i) * LL + n] = v;
      }
    }
  }
}

// ---------------------------------------------------------------------------
// Row softmax over 1024 entries; one 256-thread (8-wave) block per row.
// ---------------------------------------------------------------------------
__global__ __launch_bounds__(256) void softmax_kernel(float* __restrict__ attn) {
  __shared__ float redA[8];
  __shared__ float redB[8];
  float* p = attn + (size_t)blockIdx.x * LL;
  const int t = threadIdx.x, lane = t & 31, wid = t >> 5;

  float v[4], m = -3.4e38f;
#pragma unroll
  for (int j = 0; j < 4; ++j) { v[j] = p[t + j * 256]; m = fmaxf(m, v[j]); }
#pragma unroll
  for (int off = 16; off > 0; off >>= 1) m = fmaxf(m, __shfl_xor(m, off, 32));
  if (lane == 0) redA[wid] = m;
  __syncthreads();
  float gm = redA[0];
#pragma unroll
  for (int i = 1; i < 8; ++i) gm = fmaxf(gm, redA[i]);

  float s = 0.0f;
#pragma unroll
  for (int j = 0; j < 4; ++j) { v[j] = __expf(v[j] - gm); s += v[j]; }
#pragma unroll
  for (int off = 16; off > 0; off >>= 1) s += __shfl_xor(s, off, 32);
  if (lane == 0) redB[wid] = s;
  __syncthreads();
  float tot = 0.0f;
#pragma unroll
  for (int i = 0; i < 8; ++i) tot += redB[i];
  const float inv = 1.0f / tot;
#pragma unroll
  for (int j = 0; j < 4; ++j) p[t + j * 256] = v[j] * inv;
}

// ---------------------------------------------------------------------------
// Context: O[b,h] = P(f32,[L,L]) @ V(bf16); BM=128 BN=64, wave tile 32x32.
// P converted f32->bf16 while staging; V transposed (k-major) into LDS.
// ---------------------------------------------------------------------------
__global__ __launch_bounds__(256) void pv_kernel(
    const float* __restrict__ P, const bf16_t* __restrict__ V,
    bf16_t* __restrict__ O) {
  __shared__ alignas(16) bf16_t As[128 * LDT];
  __shared__ alignas(16) bf16_t Bs[64 * LDT];

  const int b = blockIdx.z >> 4, h = blockIdx.z & 15;
  const int m0 = blockIdx.y * 128;
  const float*  Pg = P + (size_t)(b * HH + h) * LL * LL;
  const bf16_t* Vg = V + (size_t)b * LL * DD + (size_t)h * HDIM;

  const int lane = threadIdx.x & 31, wid = threadIdx.x >> 5;
  const int wm = (wid & 3) * 32, wn = (wid >> 2) * 32;
  const int frow = threadIdx.x >> 1, fseg = (threadIdx.x & 1) << 4;
  const int vn = threadIdx.x >> 2, vks = (threadIdx.x & 3) * 8;

  v8f acc[2][2] = {};

  for (int k0 = 0; k0 < LL; k0 += 32) {
    {  // A tile: 128x32 f32 -> bf16
      const float4* g = (const float4*)(Pg + (size_t)(m0 + frow) * LL + k0 + fseg);
      float4 f0 = g[0], f1 = g[1], f2 = g[2], f3 = g[3];
      v8bf p0, p1;
      p0[0] = (bf16_t)f0.x; p0[1] = (bf16_t)f0.y; p0[2] = (bf16_t)f0.z; p0[3] = (bf16_t)f0.w;
      p0[4] = (bf16_t)f1.x; p0[5] = (bf16_t)f1.y; p0[6] = (bf16_t)f1.z; p0[7] = (bf16_t)f1.w;
      p1[0] = (bf16_t)f2.x; p1[1] = (bf16_t)f2.y; p1[2] = (bf16_t)f2.z; p1[3] = (bf16_t)f2.w;
      p1[4] = (bf16_t)f3.x; p1[5] = (bf16_t)f3.y; p1[6] = (bf16_t)f3.z; p1[7] = (bf16_t)f3.w;
      *(v8bf*)(As + frow * LDT + fseg)     = p0;
      *(v8bf*)(As + frow * LDT + fseg + 8) = p1;
    }
    {  // B tile: Bs[n][kk] = V[k0+kk][n]  (64 x 32 transpose gather)
      v8bf vb;
#pragma unroll
      for (int j = 0; j < 8; ++j) vb[j] = Vg[(size_t)(k0 + vks + j) * DD + vn];
      *(v8bf*)(Bs + vn * LDT + vks) = vb;
    }
    __syncthreads();
    const int arow = lane & 15, koff = (lane < 16) ? 0 : 8;
    const int kb = (lane < 16) ? 0 : 8;
    v16bf af[2], bq[2];
#pragma unroll
    for (int mt = 0; mt < 2; ++mt)
      af[mt] = frag_rowK(&As[(wm + mt * 16 + arow) * LDT + koff]);
#pragma unroll
    for (int nt = 0; nt < 2; ++nt)
      bq[nt] = frag_colK(&Bs[(wn + nt * 16 + (lane & 15)) * LDT + kb]);
#pragma unroll
    for (int mt = 0; mt < 2; ++mt)
#pragma unroll
      for (int nt = 0; nt < 2; ++nt)
        acc[mt][nt] = __builtin_amdgcn_wmma_f32_16x16x32_bf16(
            false, af[mt], false, bq[nt], (short)0, acc[mt][nt], false, false);
    __syncthreads();
  }

  bf16_t* Og = O + (size_t)b * LL * DD + (size_t)h * HDIM;
#pragma unroll
  for (int mt = 0; mt < 2; ++mt) {
#pragma unroll
    for (int nt = 0; nt < 2; ++nt) {
      const int n  = wn + nt * 16 + (lane & 15);
      const int mb = m0 + wm + mt * 16 + ((lane < 16) ? 0 : 8);
#pragma unroll
      for (int i = 0; i < 8; ++i)
        Og[(size_t)(mb + i) * DD + n] = (bf16_t)acc[mt][nt][i];
    }
  }
}

// ---------------------------------------------------------------------------
// Orchestration
// ---------------------------------------------------------------------------
extern "C" void kernel_launch(void* const* d_in, const int* in_sizes, int n_in,
                              void* d_out, int out_size, void* d_ws, size_t ws_size,
                              hipStream_t stream) {
  (void)in_sizes; (void)n_in; (void)out_size; (void)ws_size;

  const float* x_real = (const float*)d_in[0];
  const float* x_imag = (const float*)d_in[1];
  const int*   mask   = (const int*)d_in[2];
  const float* W[8];
  const float* Bia[8];
  for (int i = 0; i < 8; ++i) { W[i] = (const float*)d_in[3 + 2 * i]; Bia[i] = (const float*)d_in[4 + 2 * i]; }
  const float* ent = (const float*)d_in[19];

  const size_t NX = (size_t)BB * LL * DD;  // 4M elems
  const size_t NW = (size_t)DD * DD;       // 1M elems

  bf16_t* xr_bf = (bf16_t*)d_ws;
  bf16_t* xi_bf = xr_bf + NX;
  bf16_t* w_bf  = xi_bf + NX;              // 8 * NW
  bf16_t* qkv   = w_bf + 8 * NW;           // 6 * NX : qr qi kr ki vr vi
  bf16_t* o_r   = qkv + 6 * NX;
  bf16_t* o_i   = o_r + NX;

  float* out_real = (float*)d_out;
  float* out_imag = out_real + NX;
  float* attn     = out_imag + NX;

  // 1) pack everything to bf16
  pack_bf16_kernel<<<(int)(NX / 1024), 256, 0, stream>>>(x_real, xr_bf, (int)NX);
  pack_bf16_kernel<<<(int)(NX / 1024), 256, 0, stream>>>(x_imag, xi_bf, (int)NX);
  for (int i = 0; i < 8; ++i)
    pack_bf16_kernel<<<(int)(NW / 1024), 256, 0, stream>>>(W[i], w_bf + i * NW, (int)NW);

  // 2) input projections with fused noise (+ ent diag for q,k)
  dim3 gL(DD / 128, MTOK / 128);
  const bf16_t* xs[6] = { xr_bf, xi_bf, xr_bf, xi_bf, xr_bf, xi_bf };
  for (int i = 0; i < 6; ++i) {
    const int nid  = i / 2;         // 0:q 1:k 2:v
    const int imag = i & 1;
    const float* e = (nid < 2) ? ent : nullptr;
    gemm_bf16_kernel<1><<<gL, 256, 0, stream>>>(
        xs[i], w_bf + i * NW, Bia[i], MTOK, DD, DD,
        nid, imag, e, qkv + i * NX, nullptr);
  }

  // 3) interference scores -> attn region of d_out
  dim3 gS(LL / 128, LL / 128, BB * HH);
  scores_kernel<<<gS, 256, 0, stream>>>(qkv + 0 * NX, qkv + 1 * NX,
                                        qkv + 2 * NX, qkv + 3 * NX, mask, attn);

  // 4) softmax rows
  softmax_kernel<<<BB * HH * LL, 256, 0, stream>>>(attn);

  // 5) context = P @ V (real, imag)
  dim3 gP(1, LL / 128, BB * HH);
  pv_kernel<<<gP, 256, 0, stream>>>(attn, qkv + 4 * NX, o_r);
  pv_kernel<<<gP, 256, 0, stream>>>(attn, qkv + 5 * NX, o_i);

  // 6) output projections -> d_out (f32)
  gemm_bf16_kernel<0><<<gL, 256, 0, stream>>>(o_r, w_bf + 6 * NW, Bia[6], MTOK, DD, DD,
                                              0, 0, nullptr, nullptr, out_real);
  gemm_bf16_kernel<0><<<gL, 256, 0, stream>>>(o_i, w_bf + 7 * NW, Bia[7], MTOK, DD, DD,
                                              0, 0, nullptr, nullptr, out_imag);
}
